// SchNetCore_1709396984149
// MI455X (gfx1250) — compile-verified, matrix operand order checked by reading, not run on hardware
//
#include <hip/hip_runtime.h>
#include <hip/hip_bf16.h>

#define NATOMS 50000
#define NPAIRS 1600000
#define FEAT   64
#define KRBF   20
#define NLAYER 3
#define CUTF   0.5f
#define PAIRBLOCKS 1250   // persistent blocks; NPAIRS / (1250*128) = 10 iterations

typedef __attribute__((ext_vector_type(16))) __bf16 v16bf;
typedef __attribute__((ext_vector_type(8)))  float  v8f;

// K index held by element e of an A/B operand register, for wave-half `half`.
// (Two contiguous 8-element runs per lane.)
__device__ __forceinline__ int a_k_of(int e, int half) {
    return ((e < 8) ? e : e + 8) + 8 * half;
}

template <bool REUSE_A>
__device__ __forceinline__ v8f wmma_bf16(v16bf a, v16bf b, v8f c) {
    return __builtin_amdgcn_wmma_f32_16x16x32_bf16(
        false, a, false, b, (short)0, c, REUSE_A, false);
}

__device__ __forceinline__ float ssp(float v) {
    float e = __expf(v);
    float r = __logf(1.0f + e) - 0.69314718056f;
    return (v > 15.0f) ? (v - 0.69314718056f) : r;
}

// ---- Swizzled weight staging ----------------------------------------------
// dst holds ntiles (kt*4+nt) B-tiles; tile t, lane vl owns 16 contiguous bf16
// in exact WMMA B-register order. Load at GEMM time = 2x ds_load_b128.
__device__ __forceinline__ void stage_swz(const float* __restrict__ W,
                                          __bf16* dst, int ntiles, int kvalid) {
    int tid = threadIdx.x;
    int t = tid >> 5, vl = tid & 31;
    if (t < ntiles) {
        int kt = t >> 2, nt = t & 3;
        int n = nt * 16 + (vl & 15), hh = vl >> 4;
        v16bf tmp;
#pragma unroll
        for (int e = 0; e < 16; ++e) {
            int k = kt * 32 + a_k_of(e, hh);
            tmp[e] = (k < kvalid) ? (__bf16)W[k * FEAT + n] : (__bf16)0.0f;
        }
        *(v16bf*)(dst + (size_t)(t * 32 + vl) * 16) = tmp;
    }
}

__device__ __forceinline__ v16bf load_B_swz(const __bf16* wsw, int kt, int nt) {
    int lane = threadIdx.x & 31;
    return *(const v16bf*)(wsw + (size_t)((((kt << 2) | nt) * 32) + lane) * 16);
}

// One A-operand against 4 N-tiles; reuse-A hint set on the first three so the
// operand cache feeds the following identical WMMA.
__device__ __forceinline__ void gemm4(const __bf16* wsw, int kt, v16bf a, v8f* acc) {
    acc[0] = wmma_bf16<true >(a, load_B_swz(wsw, kt, 0), acc[0]);
    acc[1] = wmma_bf16<true >(a, load_B_swz(wsw, kt, 1), acc[1]);
    acc[2] = wmma_bf16<true >(a, load_B_swz(wsw, kt, 2), acc[2]);
    acc[3] = wmma_bf16<false>(a, load_B_swz(wsw, kt, 3), acc[3]);
}

// A-tile (16x32) straight from a row-major f32 matrix: 4x b128 + cvt.
__device__ __forceinline__ v16bf load_A_gmem(const float* __restrict__ g,
                                             long rowBase, int kt) {
    int lane = threadIdx.x & 31;
    int m = lane & 15, half = lane >> 4;
    const float* row = g + (rowBase + m) * FEAT + kt * 32 + 8 * half;
    float4 f0 = *(const float4*)(row);
    float4 f1 = *(const float4*)(row + 4);
    float4 f2 = *(const float4*)(row + 16);
    float4 f3 = *(const float4*)(row + 20);
    v16bf a;
    a[0] = (__bf16)f0.x;  a[1] = (__bf16)f0.y;  a[2]  = (__bf16)f0.z;  a[3]  = (__bf16)f0.w;
    a[4] = (__bf16)f1.x;  a[5] = (__bf16)f1.y;  a[6]  = (__bf16)f1.z;  a[7]  = (__bf16)f1.w;
    a[8] = (__bf16)f2.x;  a[9] = (__bf16)f2.y;  a[10] = (__bf16)f2.z;  a[11] = (__bf16)f2.w;
    a[12] = (__bf16)f3.x; a[13] = (__bf16)f3.y; a[14] = (__bf16)f3.z;  a[15] = (__bf16)f3.w;
    return a;
}

// Store a D column (8 rows) into the swizzled A-layout buffer for the next GEMM.
__device__ __forceinline__ void t1_store_col(__bf16* t1w, int col, int half,
                                             const float* vals /*8 activated*/) {
    int kt2 = col >> 5, kk = col & 31, hh = (kk >> 3) & 1;
    int kk2 = kk - 8 * hh;
    int e   = (kk2 >= 16) ? kk2 - 8 : kk2;
    __bf16* tp = t1w + (size_t)(kt2 * 32 + 16 * hh) * 16 + e;
#pragma unroll
    for (int r = 0; r < 8; ++r)
        tp[(size_t)(r + 8 * half) * 16] = (__bf16)vals[r];
}

// ---- Kernels ---------------------------------------------------------------

__global__ void embed_kernel(const int* __restrict__ z,
                             const float* __restrict__ emb,
                             float* __restrict__ x) {
    long i = (long)blockIdx.x * blockDim.x + threadIdx.x;
    if (i >= (long)NATOMS * FEAT) return;
    int n = (int)(i >> 6), f = (int)(i & 63);
    x[i] = emb[z[n] * FEAT + f];
}

// h = x @ W
__global__ __launch_bounds__(256) void node_in_kernel(
    const float* __restrict__ x, const float* __restrict__ W,
    float* __restrict__ h) {
    __shared__ __attribute__((aligned(32))) __bf16 wsw[8 * 32 * 16];
    stage_swz(W, wsw, 8, 64);
    __syncthreads();

    int tid = threadIdx.x, wave = tid >> 5, lane = tid & 31;
    int m = lane & 15, half = lane >> 4;
    long rowBase = ((long)blockIdx.x * 8 + wave) * 16;
    if (rowBase >= NATOMS) return;

    v8f acc[4] = {};
#pragma unroll
    for (int kt = 0; kt < 2; ++kt)
        gemm4(wsw, kt, load_A_gmem(x, rowBase, kt), acc);
#pragma unroll
    for (int r = 0; r < 8; ++r) {
        long row = rowBase + r + 8 * half;
#pragma unroll
        for (int nt = 0; nt < 4; ++nt)
            h[row * FEAT + nt * 16 + m] = acc[nt][r];
    }
}

// Persistent fused per-pair pipeline: RBF -> WMMA -> ssp -> WMMA -> cutoff
//   -> gather h[idx_j] -> multiply -> hardware f32 atomic scatter into agg.
// Weights staged into LDS once per block, reused across grid-stride iterations.
__global__ __launch_bounds__(256) void pair_kernel(
    const int* __restrict__ pair_i, const int* __restrict__ pair_j,
    const float* __restrict__ d_ij,
    const float* __restrict__ Wf1, const float* __restrict__ bf1,
    const float* __restrict__ Wf2, const float* __restrict__ bf2,
    const float* __restrict__ h, float* __restrict__ agg) {
    __shared__ __attribute__((aligned(32))) __bf16 wf1sw[4 * 32 * 16]; // K padded 20->32
    __shared__ __attribute__((aligned(32))) __bf16 wf2sw[8 * 32 * 16];
    __shared__ __attribute__((aligned(32))) __bf16 t1sw[8][2 * 32 * 16];

    stage_swz(Wf1, wf1sw, 4, KRBF);
    stage_swz(Wf2, wf2sw, 8, 64);
    __syncthreads();

    int tid = threadIdx.x, wave = tid >> 5, lane = tid & 31;
    int m = lane & 15, half = lane >> 4;
    __bf16* t1w = t1sw[wave];

    const float delta  = CUTF / (float)(KRBF - 1);
    const float inv2d2 = 0.5f / (delta * delta);

    float bias1[4], bias2[4];
#pragma unroll
    for (int nt = 0; nt < 4; ++nt) {
        bias1[nt] = bf1[nt * 16 + m];
        bias2[nt] = bf2[nt * 16 + m];
    }

    const long step = (long)gridDim.x * 128;  // 8 waves * 16 pairs per block
    for (long tileBase = ((long)blockIdx.x * 8 + wave) * 16; tileBase < NPAIRS;
         tileBase += step) {
        // prefetch next iteration's pair metadata (global_prefetch_b8)
        long nb = tileBase + step;
        if (nb < NPAIRS) {
            __builtin_prefetch(pair_i + nb, 0, 0);
            __builtin_prefetch(pair_j + nb, 0, 0);
            __builtin_prefetch(d_ij + nb, 0, 0);
        }

        // pair metadata as wide vector loads (rows r+8*half belong to this lane)
        float d = d_ij[tileBase + m];
        const long mb = tileBase + 8 * half;
        int4   ia = *(const int4*)(pair_i + mb);
        int4   ib = *(const int4*)(pair_i + mb + 4);
        int4   ja = *(const int4*)(pair_j + mb);
        int4   jb = *(const int4*)(pair_j + mb + 4);
        float4 da = *(const float4*)(d_ij + mb);
        float4 db = *(const float4*)(d_ij + mb + 4);
        int   ii[8] = {ia.x, ia.y, ia.z, ia.w, ib.x, ib.y, ib.z, ib.w};
        int   jj[8] = {ja.x, ja.y, ja.z, ja.w, jb.x, jb.y, jb.z, jb.w};
        float dv[8] = {da.x, da.y, da.z, da.w, db.x, db.y, db.z, db.w};

        // A-tile: gaussian RBF expansion built directly in register layout
        v16bf afa;
#pragma unroll
        for (int e = 0; e < 16; ++e) {
            int k = a_k_of(e, half);
            float v = 0.0f;
            if (k < KRBF) { float t = d - (float)k * delta; v = __expf(-t * t * inv2d2); }
            afa[e] = (__bf16)v;
        }

        // GEMM1 (16x32 @ 32x64) -> +bf1 -> ssp -> swizzled LDS store (A-order)
        v8f c1[4] = {};
        gemm4(wf1sw, 0, afa, c1);
#pragma unroll
        for (int nt = 0; nt < 4; ++nt) {
            float av[8];
#pragma unroll
            for (int r = 0; r < 8; ++r) av[r] = ssp(c1[nt][r] + bias1[nt]);
            t1_store_col(t1w, nt * 16 + m, half, av);
        }
        asm volatile("" ::: "memory"); // keep DS store->load order (in-order per wave)

        // gather h[idx_j] rows early to overlap with the WMMA chain below
        float hv[8][4];
#pragma unroll
        for (int r = 0; r < 8; ++r) {
            const float* hrow = h + (long)jj[r] * FEAT + m;
#pragma unroll
            for (int nt = 0; nt < 4; ++nt) hv[r][nt] = hrow[nt * 16];
        }

        // GEMM2 (16x64 @ 64x64): A reload = 2x ds_load_b128 per k-tile
        v8f acc[4] = {};
#pragma unroll
        for (int kt = 0; kt < 2; ++kt) {
            v16bf a = *(const v16bf*)(t1w + (size_t)(kt * 32 + lane) * 16);
            gemm4(wf2sw, kt, a, acc);
        }

        // cutoff, multiply, atomic scatter into L2-resident agg
#pragma unroll
        for (int r = 0; r < 8; ++r) {
            float dd = dv[r];
            float fc = (dd < CUTF) ? 0.5f * (__cosf(6.28318530718f * dd) + 1.0f) : 0.0f;
            float* arow = agg + (long)ii[r] * FEAT + m;
#pragma unroll
            for (int nt = 0; nt < 4; ++nt) {
                float wij = (acc[nt][r] + bias2[nt]) * fc;
                unsafeAtomicAdd(&arow[nt * 16], hv[r][nt] * wij);
            }
        }
    }
}

// x += ssp(agg @ Wo1 + bo1) @ Wo2 + bo2
__global__ __launch_bounds__(256) void node_update_kernel(
    const float* __restrict__ agg,
    const float* __restrict__ Wo1, const float* __restrict__ bo1,
    const float* __restrict__ Wo2, const float* __restrict__ bo2,
    float* __restrict__ x) {
    __shared__ __attribute__((aligned(32))) __bf16 w1sw[8 * 32 * 16];
    __shared__ __attribute__((aligned(32))) __bf16 w2sw[8 * 32 * 16];
    __shared__ __attribute__((aligned(32))) __bf16 t1sw[8][2 * 32 * 16];
    stage_swz(Wo1, w1sw, 8, 64);
    stage_swz(Wo2, w2sw, 8, 64);
    __syncthreads();

    int tid = threadIdx.x, wave = tid >> 5, lane = tid & 31;
    int m = lane & 15, half = lane >> 4;
    long rowBase = ((long)blockIdx.x * 8 + wave) * 16;
    if (rowBase >= NATOMS) return;

    v8f acc[4] = {};
#pragma unroll
    for (int kt = 0; kt < 2; ++kt)
        gemm4(w1sw, kt, load_A_gmem(agg, rowBase, kt), acc);

    __bf16* t1w = t1sw[wave];
#pragma unroll
    for (int nt = 0; nt < 4; ++nt) {
        int col = nt * 16 + m;
        float bias = bo1[col];
        float av[8];
#pragma unroll
        for (int r = 0; r < 8; ++r) av[r] = ssp(acc[nt][r] + bias);
        t1_store_col(t1w, col, half, av);
    }
    asm volatile("" ::: "memory");

    v8f acc2[4] = {};
#pragma unroll
    for (int kt = 0; kt < 2; ++kt) {
        v16bf a = *(const v16bf*)(t1w + (size_t)(kt * 32 + lane) * 16);
        gemm4(w2sw, kt, a, acc2);
    }
#pragma unroll
    for (int r = 0; r < 8; ++r) {
        long row = rowBase + r + 8 * half;
#pragma unroll
        for (int nt = 0; nt < 4; ++nt) {
            int col = nt * 16 + m;
            x[row * FEAT + col] += acc2[nt][r] + bo2[col];
        }
    }
}

// energy = ssp(x @ Wp1 + bp1) @ Wp2 + bp2  (matvec via wave shuffle reduction)
__global__ __launch_bounds__(256) void readout_kernel(
    const float* __restrict__ x,
    const float* __restrict__ Wp1, const float* __restrict__ bp1,
    const float* __restrict__ Wp2, const float* __restrict__ bp2,
    float* __restrict__ out) {
    __shared__ __attribute__((aligned(32))) __bf16 w1sw[8 * 32 * 16];
    stage_swz(Wp1, w1sw, 8, 64);
    __syncthreads();

    int tid = threadIdx.x, wave = tid >> 5, lane = tid & 31;
    int m = lane & 15, half = lane >> 4;
    long rowBase = ((long)blockIdx.x * 8 + wave) * 16;
    if (rowBase >= NATOMS) return;

    v8f acc[4] = {};
#pragma unroll
    for (int kt = 0; kt < 2; ++kt)
        gemm4(w1sw, kt, load_A_gmem(x, rowBase, kt), acc);

    float wp2v[4], b1v[4];
#pragma unroll
    for (int nt = 0; nt < 4; ++nt) {
        int col = nt * 16 + m;
        wp2v[nt] = Wp2[col];
        b1v[nt]  = bp1[col];
    }
    float bp2v = bp2[0];

#pragma unroll
    for (int r = 0; r < 8; ++r) {
        float p = 0.0f;
#pragma unroll
        for (int nt = 0; nt < 4; ++nt)
            p += ssp(acc[nt][r] + b1v[nt]) * wp2v[nt];
        p += __shfl_xor(p, 1);
        p += __shfl_xor(p, 2);
        p += __shfl_xor(p, 4);
        p += __shfl_xor(p, 8);
        if (m == 0) out[rowBase + r + 8 * half] = p + bp2v;
    }
}

// ---- Host driver -----------------------------------------------------------

extern "C" void kernel_launch(void* const* d_in, const int* in_sizes, int n_in,
                              void* d_out, int out_size, void* d_ws, size_t ws_size,
                              hipStream_t stream) {
    const int*   atomic_numbers = (const int*)d_in[0];
    const int*   pair_indices   = (const int*)d_in[1];
    const float* dij            = (const float*)d_in[2];
    const float* emb            = (const float*)d_in[3];
    const float* W_in           = (const float*)d_in[4];
    const float* Wf1            = (const float*)d_in[5];
    const float* bf1            = (const float*)d_in[6];
    const float* Wf2            = (const float*)d_in[7];
    const float* bf2            = (const float*)d_in[8];
    const float* Wo1            = (const float*)d_in[9];
    const float* bo1            = (const float*)d_in[10];
    const float* Wo2            = (const float*)d_in[11];
    const float* bo2            = (const float*)d_in[12];
    const float* Wp1            = (const float*)d_in[13];
    const float* bp1            = (const float*)d_in[14];
    const float* Wp2            = (const float*)d_in[15];
    const float* bp2            = (const float*)d_in[16];
    float* out = (float*)d_out;

    float* x   = (float*)d_ws;
    float* h   = x + (size_t)NATOMS * FEAT;
    float* agg = h + (size_t)NATOMS * FEAT;

    const int* pair_i = pair_indices;
    const int* pair_j = pair_indices + NPAIRS;

    embed_kernel<<<(NATOMS * FEAT + 255) / 256, 256, 0, stream>>>(atomic_numbers, emb, x);

    int nodeBlocks = (NATOMS / 16 + 7) / 8;   // 391

    for (int l = 0; l < NLAYER; ++l) {
        node_in_kernel<<<nodeBlocks, 256, 0, stream>>>(x, W_in + (size_t)l * 64 * 64, h);
        hipMemsetAsync(agg, 0, (size_t)NATOMS * FEAT * sizeof(float), stream);
        pair_kernel<<<PAIRBLOCKS, 256, 0, stream>>>(
            pair_i, pair_j, dij,
            Wf1 + (size_t)l * KRBF * 64, bf1 + (size_t)l * 64,
            Wf2 + (size_t)l * 64 * 64,   bf2 + (size_t)l * 64,
            h, agg);
        node_update_kernel<<<nodeBlocks, 256, 0, stream>>>(
            agg, Wo1 + (size_t)l * 64 * 64, bo1 + (size_t)l * 64,
            Wo2 + (size_t)l * 64 * 64, bo2 + (size_t)l * 64, x);
    }

    readout_kernel<<<nodeBlocks, 256, 0, stream>>>(x, Wp1, bp1, Wp2, bp2, out);
}
